// NGCF_42348377538882
// MI455X (gfx1250) — compile-verified, matrix operand order checked by reading, not run on hardware
//
#include <hip/hip_runtime.h>
#include <hip/hip_bf16.h>
#include <math.h>

// ---------------------------------------------------------------------------
// NGCF forward for MI455X (gfx1250, wave32, WMMA).
//   N = 150000 nodes, D = 64, 3 layers, NNZ = 2.4M, B = 4096.
//   SpMM: L2-resident gather + fp32 HW atomics (memory bound).
//   Dense (LE+E)@W1 + (LE*E)@W2: V_WMMA_F32_16X16X32_F16, f32 accum,
//   fused bias + leaky-relu + row-normalization epilogue.
// ---------------------------------------------------------------------------

typedef __attribute__((ext_vector_type(16))) _Float16 v16h;
typedef __attribute__((ext_vector_type(8)))  float    v8f;

#define NGCF_NUSER 50000
#define NGCF_NITEM 100000
#define NGCF_N     150000
#define NGCF_D     64
#define NGCF_NNZ   2400000
#define NGCF_B     4096
#define NEG_SLOPE  0.2f
#define NGCF_EPS   1e-12f

// ---------------------------------------------------------------------------
// SpMM: LE[row] += val * E[col] for each edge. 16 lanes per edge, float4 each.
// E/LE are 38.4 MB -> L2 resident; atomics are hardware global_atomic_add_f32.
// ---------------------------------------------------------------------------
__global__ __launch_bounds__(256) void ngcf_spmm(
    const int* __restrict__ rows, const int* __restrict__ cols,
    const float* __restrict__ vals, const float* __restrict__ E,
    float* __restrict__ LE, int nnz)
{
    long long gid = (long long)blockIdx.x * blockDim.x + threadIdx.x;
    int edge = (int)(gid >> 4);
    int seg  = (int)(gid & 15);
    if (edge >= nnz) return;
    int r = rows[edge];
    int c = cols[edge];
    float v = vals[edge];
    const float4 x = *(const float4*)(E + (size_t)c * NGCF_D + seg * 4);
    float* dst = LE + (size_t)r * NGCF_D + seg * 4;
    unsafeAtomicAdd(dst + 0, v * x.x);
    unsafeAtomicAdd(dst + 1, v * x.y);
    unsafeAtomicAdd(dst + 2, v * x.z);
    unsafeAtomicAdd(dst + 3, v * x.w);
}

// ---------------------------------------------------------------------------
// Fused dense layer:
//   H = (LE+E) @ W1 + (LE*E) @ W2 + (b1+b2)
//   Enext = leaky_relu(H);  Lnorm = Enext / max(||row||, eps)
// One wave per 16-row tile; 4 col-tiles x 2 k-steps x 2 weights = 16 WMMAs.
// W1^T/W2^T staged in LDS as f16 once per workgroup (8 waves).
// ---------------------------------------------------------------------------
__global__ __launch_bounds__(256) void ngcf_gemm_wmma(
    const float* __restrict__ Esrc, const float* __restrict__ LE,
    const float* __restrict__ W1,  const float* __restrict__ W2,
    const float* __restrict__ b1,  const float* __restrict__ b2,
    float* __restrict__ Edst, float* __restrict__ Lnorm, int nTiles)
{
    __shared__ _Float16 sW1T[NGCF_D * NGCF_D];   // transposed: [n][k]
    __shared__ _Float16 sW2T[NGCF_D * NGCF_D];
    __shared__ float    sBias[NGCF_D];

    const int tid = threadIdx.x;
    for (int i = tid; i < NGCF_D * NGCF_D; i += 256) {
        int k = i >> 6, n = i & 63;
        sW1T[n * NGCF_D + k] = (_Float16)W1[i];
        sW2T[n * NGCF_D + k] = (_Float16)W2[i];
    }
    if (tid < NGCF_D) sBias[tid] = b1[tid] + b2[tid];
    __syncthreads();

    const int wave = tid >> 5;
    const int lane = tid & 31;
    const int tile = blockIdx.x * 8 + wave;
    if (tile >= nTiles) return;          // wave-uniform: EXEC all-1s for WMMA

    const int mload = lane & 15;         // A-operand row held by this lane
    const int grp   = lane >> 4;         // half-wave (selects K sub-block)
    const float* leRow = LE   + (size_t)(tile * 16 + mload) * NGCF_D;
    const float* eRow  = Esrc + (size_t)(tile * 16 + mload) * NGCF_D;

    v8f acc[4] = {};                     // 4 column tiles of 16

    #pragma unroll
    for (int ks = 0; ks < 2; ++ks) {     // K = 64 = 2 x 32
        // Build both A operands once (ISA 16-bit A layout, p.108)
        v16h a1, a2;
        #pragma unroll
        for (int v = 0; v < 8; ++v) {
            int k0 = ks * 32 + ((v >> 2) << 4) + (grp << 3) + ((v & 3) << 1);
            float le0 = leRow[k0], le1 = leRow[k0 + 1];
            float e0  = eRow[k0],  e1  = eRow[k0 + 1];
            a1[2 * v]     = (_Float16)(le0 + e0);
            a1[2 * v + 1] = (_Float16)(le1 + e1);
            a2[2 * v]     = (_Float16)(le0 * e0);
            a2[2 * v + 1] = (_Float16)(le1 * e1);
        }
        #pragma unroll
        for (int t = 0; t < 4; ++t) {
            int ncol = t * 16 + (lane & 15);      // B-operand column
            v16h bw1, bw2;
            #pragma unroll
            for (int v = 0; v < 8; ++v) {
                int k0 = ks * 32 + ((v >> 2) << 4) + (grp << 3) + ((v & 3) << 1);
                bw1[2 * v]     = sW1T[ncol * NGCF_D + k0];
                bw1[2 * v + 1] = sW1T[ncol * NGCF_D + k0 + 1];
                bw2[2 * v]     = sW2T[ncol * NGCF_D + k0];
                bw2[2 * v + 1] = sW2T[ncol * NGCF_D + k0 + 1];
            }
            acc[t] = __builtin_amdgcn_wmma_f32_16x16x32_f16(
                false, a1, false, bw1, (short)0, acc[t], false, false);
            acc[t] = __builtin_amdgcn_wmma_f32_16x16x32_f16(
                false, a2, false, bw2, (short)0, acc[t], false, false);
        }
    }

    // Epilogue on the C layout: value (t,v) is H[v + 8*grp][t*16 + (lane&15)]
    float vals[4][8];
    #pragma unroll
    for (int t = 0; t < 4; ++t) {
        float bsum = sBias[t * 16 + (lane & 15)];
        #pragma unroll
        for (int v = 0; v < 8; ++v) {
            float h = acc[t][v] + bsum;
            vals[t][v] = (h >= 0.0f) ? h : NEG_SLOPE * h;
        }
    }
    // Row L2 norm: reduce squares over the 16 lanes of each half-wave.
    float nrm[8];
    #pragma unroll
    for (int v = 0; v < 8; ++v) {
        float s = 0.0f;
        #pragma unroll
        for (int t = 0; t < 4; ++t) s += vals[t][v] * vals[t][v];
        s += __shfl_xor(s, 1, 32);   // xor masks 1..8 stay within the
        s += __shfl_xor(s, 2, 32);   // 16-lane half (never flip bit 4)
        s += __shfl_xor(s, 4, 32);
        s += __shfl_xor(s, 8, 32);
        nrm[v] = fmaxf(sqrtf(s), NGCF_EPS);
    }
    #pragma unroll
    for (int t = 0; t < 4; ++t) {
        #pragma unroll
        for (int v = 0; v < 8; ++v) {
            int mrow = v + 8 * grp;
            int n    = t * 16 + (lane & 15);
            size_t off = (size_t)(tile * 16 + mrow) * NGCF_D + n;
            Edst[off]  = vals[t][v];
            Lnorm[off] = vals[t][v] / nrm[v];
        }
    }
}

// ---------------------------------------------------------------------------
// Final gather: out = [u_emb | pos | neg], each 4096 x 256 (4 layers x 64).
// ---------------------------------------------------------------------------
__global__ __launch_bounds__(256) void ngcf_gather(
    const int* __restrict__ users, const int* __restrict__ pos,
    const int* __restrict__ neg,
    const float* __restrict__ E0, const float* __restrict__ L1,
    const float* __restrict__ L2, const float* __restrict__ L3,
    float* __restrict__ out)
{
    int gid = blockIdx.x * blockDim.x + threadIdx.x;
    int row = gid >> 6;                 // 64 float4 chunks per 256-wide row
    int c4  = (gid & 63) << 2;
    if (row >= 3 * NGCF_B) return;
    int which = row >> 12;
    int b     = row & (NGCF_B - 1);
    int node;
    if (which == 0)      node = users[b] - 1;
    else if (which == 1) node = NGCF_NUSER + pos[b] - 1;
    else                 node = NGCF_NUSER + neg[b] - 1;
    int layer = c4 >> 6;
    int lc    = c4 & 63;
    const float* src = (layer == 0) ? E0 : (layer == 1) ? L1
                     : (layer == 2) ? L2 : L3;
    float4 x = *(const float4*)(src + (size_t)node * NGCF_D + lc);
    *(float4*)(out + (size_t)row * 256 + c4) = x;
}

// ---------------------------------------------------------------------------
extern "C" void kernel_launch(void* const* d_in, const int* in_sizes, int n_in,
                              void* d_out, int out_size, void* d_ws, size_t ws_size,
                              hipStream_t stream)
{
    (void)in_sizes; (void)n_in; (void)out_size; (void)ws_size;
    const float* user_emb  = (const float*)d_in[0];
    const float* item_emb  = (const float*)d_in[1];
    const float* edge_vals = (const float*)d_in[2];
    const float* W1        = (const float*)d_in[3];
    const float* b1        = (const float*)d_in[4];
    const float* W2        = (const float*)d_in[5];
    const float* b2        = (const float*)d_in[6];
    const int*   eidx      = (const int*)d_in[7];
    const int*   users     = (const int*)d_in[8];
    const int*   pos       = (const int*)d_in[9];
    const int*   neg       = (const int*)d_in[10];
    float* out = (float*)d_out;

    const size_t ND = (size_t)NGCF_N * NGCF_D;
    float* ws   = (float*)d_ws;
    float* E0   = ws;                 // layer-0 slice of all_E (unnormalized)
    float* Ecur = ws + 1 * ND;        // running (unnormalized) embeddings
    float* LE   = ws + 2 * ND;        // SpMM accumulator
    float* Ln0  = ws + 3 * ND;        // normalized layer outputs
    float* Ln1  = ws + 4 * ND;
    float* Ln2  = ws + 5 * ND;
    float* Ln[3] = { Ln0, Ln1, Ln2 };

    // E0 = concat(user_emb, item_emb)
    hipMemcpyAsync(E0, user_emb, (size_t)NGCF_NUSER * NGCF_D * sizeof(float),
                   hipMemcpyDeviceToDevice, stream);
    hipMemcpyAsync(E0 + (size_t)NGCF_NUSER * NGCF_D, item_emb,
                   (size_t)NGCF_NITEM * NGCF_D * sizeof(float),
                   hipMemcpyDeviceToDevice, stream);

    const int* rows = eidx;
    const int* cols = eidx + NGCF_NNZ;
    const int nTiles = NGCF_N / 16;   // 9375 exactly

    for (int l = 0; l < 3; ++l) {
        const float* Esrc = (l == 0) ? E0 : Ecur;
        hipMemsetAsync(LE, 0, ND * sizeof(float), stream);
        int spmmThreads = NGCF_NNZ * 16;
        ngcf_spmm<<<dim3((spmmThreads + 255) / 256), dim3(256), 0, stream>>>(
            rows, cols, edge_vals, Esrc, LE, NGCF_NNZ);
        ngcf_gemm_wmma<<<dim3((nTiles + 7) / 8), dim3(256), 0, stream>>>(
            Esrc, LE, W1 + l * NGCF_D * NGCF_D, W2 + l * NGCF_D * NGCF_D,
            b1 + l * NGCF_D, b2 + l * NGCF_D, Ecur, Ln[l], nTiles);
    }

    int gatherThreads = 3 * NGCF_B * 64;
    ngcf_gather<<<dim3((gatherThreads + 255) / 256), dim3(256), 0, stream>>>(
        users, pos, neg, E0, Ln0, Ln1, Ln2, out);
}
